// ConditionalAffineCouplingLayer_55379308315280
// MI455X (gfx1250) — matrix operand order repaired and checked by reading, not compiled, not run on hardware
//
#include <hip/hip_runtime.h>
#include <hip/hip_bf16.h>

// ---------------------------------------------------------------------------
// ConditionalAffineCouplingLayer for MI455X (gfx1250, wave32, WMMA).
// B=131072, D=96, C=32, H=128. GEMMs run in f16 WMMA (v_wmma_f32_16x16x32_f16)
// with f32 accumulation; batch-norm statistics are deterministic two-stage
// tree reductions. Workspace usage ~213 MB (see offsets below).
// ---------------------------------------------------------------------------

typedef __attribute__((ext_vector_type(16))) _Float16 v16h;
typedef __attribute__((ext_vector_type(8)))  _Float16 v8h;
typedef __attribute__((ext_vector_type(8)))  float    v8f;
typedef __attribute__((ext_vector_type(4)))  float    v4f;

#define BATCH   131072
#define DD      96
#define CC      32
#define HH      128
#define OUTW    48          // D/2
#define K0PAD   96          // 80 padded to multiple of 32
#define SFc     2.0f
#define PI_F    3.14159265358979323846f
#define GEMM_GRID (BATCH/64)   // 2048 workgroups, 64 rows each

// ------------------------- workspace layout (bytes) -------------------------
static constexpr size_t WT_ELEMS   = 4*128*96 + 4*3*128*128 + 4*48*128; // 270336 f16
static constexpr size_t OFF_WT     = 0;                                  // 540,672 B
static constexpr size_t OFF_IN16   = 1u << 20;                           // f16 [B][96]
static constexpr size_t SZ_IN16    = (size_t)BATCH*96*2;                 // 25,165,824
static constexpr size_t OFF_HA     = OFF_IN16 + SZ_IN16;                 // f32 [B][128]
static constexpr size_t SZ_H       = (size_t)BATCH*128*4;                // 67,108,864
static constexpr size_t OFF_HB     = OFF_HA + SZ_H;
static constexpr size_t OFF_S      = OFF_HB + SZ_H;                      // f32 [B][48]
static constexpr size_t SZ_ST      = (size_t)BATCH*48*4;                 // 25,165,824
static constexpr size_t OFF_T      = OFF_S + SZ_ST;
static constexpr size_t OFF_PART   = OFF_T + SZ_ST;                      // 2048*256*4
static constexpr size_t OFF_SS     = OFF_PART + (size_t)GEMM_GRID*256*4; // scale/shift 2*128 f32
static constexpr size_t OFF_LD1    = OFF_SS + 1024;                      // 49152 f32
static constexpr size_t OFF_LD2    = OFF_LD1 + 49152*4;                  // 24576 f32

// ------------------------ weight prep: f32 -> f16 [N][K] --------------------
// W0t[net]    : [128][96]  (k>=80 zero-padded)    at f16 index net*12288
// Wst[net][j] : [128][128]                        at 49152 + (net*3+j)*16384
// Wft[net]    : [48][128]                         at 245760 + net*6144
__global__ __launch_bounds__(256)
void prep_weights_kernel(const float* __restrict__ W0,
                         const float* __restrict__ Ws,
                         const float* __restrict__ Wf,
                         _Float16* __restrict__ wt)
{
    int idx = blockIdx.x * 256 + threadIdx.x;
    if (idx >= (int)WT_ELEMS) return;
    float v;
    if (idx < 49152) {                       // W0: (4, 80, 128)
        int net = idx / 12288, rem = idx % 12288;
        int c = rem / 96, k = rem % 96;
        v = (k < 80) ? W0[(size_t)net*10240 + (size_t)k*128 + c] : 0.0f;
    } else if (idx < 245760) {               // Ws: (4, 3, 128, 128)
        int q = idx - 49152;
        int li = q / 16384, rem = q % 16384;
        int c = rem / 128, k = rem % 128;
        v = Ws[(size_t)li*16384 + (size_t)k*128 + c];
    } else {                                 // Wf: (4, 128, 48)
        int q = idx - 245760;
        int net = q / 6144, rem = q % 6144;
        int c = rem / 128, k = rem % 128;
        v = Wf[(size_t)net*6144 + (size_t)k*48 + c];
    }
    wt[idx] = (_Float16)v;
}

// --------------------------- build in1 (f16, stride 96) ---------------------
__global__ __launch_bounds__(256)
void build_in1_kernel(const float* __restrict__ x, const float* __restrict__ cond,
                      _Float16* __restrict__ in16)
{
    size_t idx = (size_t)blockIdx.x * 256 + threadIdx.x;   // B*96 total
    int row = (int)(idx / 96), j = (int)(idx % 96);
    float v;
    if (j < 48)        v = x[(size_t)row*DD + 2*j] + sinf((float)(2*j));
    else if (j < 80)   v = cond[(size_t)row*CC + (j - 48)];
    else               v = 0.0f;
    in16[idx] = (_Float16)v;
}

// ----------------------------- fused GEMM kernel ----------------------------
// 128 threads = 4 waves; each wave computes a 16-row strip x full N.
// BN_IN: input is raw f32 h of previous layer; apply scale/shift + leaky-ReLU
//        while converting to f16. Otherwise input is pre-built f16 (stride K).
// STATS: emit per-column sum/sum^2 partials (deterministic) for the next BN.
template<int K, int N, bool BN_IN, bool STATS>
__global__ __launch_bounds__(128)
void gemm_bn_kernel(const void*  __restrict__ Ain,
                    const _Float16* __restrict__ Wt,       // [N][K] f16
                    const float* __restrict__ bias,        // [N]
                    const float* __restrict__ scaleShift,  // [2*K] when BN_IN
                    float* __restrict__ Hout,              // [B][N] f32
                    float* __restrict__ partials)          // [grid][2*N]
{
    constexpr int NT = N / 16;
    constexpr int KC = K / 32;
    __shared__ alignas(32) _Float16 sW[N * K];
    __shared__ float sBias[N];
    __shared__ float sScale[BN_IN ? K : 1];
    __shared__ float sShift[BN_IN ? K : 1];
    __shared__ float sSum[STATS ? 4 * N : 1];
    __shared__ float sSq [STATS ? 4 * N : 1];

    const int tid = threadIdx.x;

    // stage weights + per-column params in LDS
    {
        const uint4* src = reinterpret_cast<const uint4*>(Wt);
        uint4* dst = reinterpret_cast<uint4*>(sW);
        for (int i = tid; i < N * K / 8; i += 128) dst[i] = src[i];
        for (int i = tid; i < N; i += 128) sBias[i] = bias[i];
        if constexpr (BN_IN) {
            for (int i = tid; i < K; i += 128) {
                sScale[i] = scaleShift[i];
                sShift[i] = scaleShift[K + i];
            }
        }
    }
    __syncthreads();

    const int wave = tid >> 5, lane = tid & 31;
    const int half = lane >> 4, l15 = lane & 15;
    const int rowBase = blockIdx.x * 64 + wave * 16;
    const size_t arow = (size_t)(rowBase + l15);

    v8f acc[NT];
#pragma unroll
    for (int nt = 0; nt < NT; ++nt)
#pragma unroll
        for (int e = 0; e < 8; ++e) acc[nt][e] = 0.0f;

#pragma unroll
    for (int kc = 0; kc < KC; ++kc) {
        // A fragment: 16x32 f16; lane l15 holds row M=l15;
        // half 0: K = kc*32 + {0..15}; half 1: +16.
        v16h a;
        const int k0 = kc * 32 + half * 16;
        if constexpr (!BN_IN) {
            const _Float16* ap = (const _Float16*)Ain + arow * K + k0;
            v8h x0 = *reinterpret_cast<const v8h*>(ap);
            v8h x1 = *reinterpret_cast<const v8h*>(ap + 8);
#pragma unroll
            for (int e = 0; e < 8; ++e) { a[e] = x0[e]; a[8 + e] = x1[e]; }
        } else {
            const float* hp = (const float*)Ain + arow * K + k0;
            v4f f0 = *reinterpret_cast<const v4f*>(hp);
            v4f f1 = *reinterpret_cast<const v4f*>(hp + 4);
            v4f f2 = *reinterpret_cast<const v4f*>(hp + 8);
            v4f f3 = *reinterpret_cast<const v4f*>(hp + 12);
#pragma unroll
            for (int e = 0; e < 8; ++e) {
                float raw = (e < 4) ? f0[e & 3] : f1[e & 3];
                float v = raw * sScale[k0 + e] + sShift[k0 + e];
                v = (v > 0.0f) ? v : 0.01f * v;            // leaky_relu
                a[e] = (_Float16)v;
                float raw2 = (e < 4) ? f2[e & 3] : f3[e & 3];
                float v2 = raw2 * sScale[k0 + 8 + e] + sShift[k0 + 8 + e];
                v2 = (v2 > 0.0f) ? v2 : 0.01f * v2;
                a[8 + e] = (_Float16)v2;
            }
        }
#pragma unroll
        for (int nt = 0; nt < NT; ++nt) {
            // B fragment: lane = column nt*16+l15; K run = kc*32 + half*16 + 0..15
            const v16h b = *reinterpret_cast<const v16h*>(
                &sW[(size_t)(nt * 16 + l15) * K + kc * 32 + half * 16]);
            acc[nt] = __builtin_amdgcn_wmma_f32_16x16x32_f16(
                false, a, false, b, (short)0, acc[nt], false, false);
        }
    }

    // epilogue: bias, store raw f32 h, deterministic column partial sums
#pragma unroll
    for (int nt = 0; nt < NT; ++nt) {
        const int c = nt * 16 + l15;
        const float bcol = sBias[c];
        float s = 0.0f, q = 0.0f;
        float* op = Hout + (size_t)(rowBase + half * 8) * N + c;
#pragma unroll
        for (int v = 0; v < 8; ++v) {        // D layout: row = v + half*8
            float val = acc[nt][v] + bcol;
            op[(size_t)v * N] = val;
            if constexpr (STATS) { s += val; q += val * val; }
        }
        if constexpr (STATS) {
            s += __shfl_xor(s, 16, 32);       // combine the two half-wave row sets
            q += __shfl_xor(q, 16, 32);
            if (half == 0) { sSum[wave * N + c] = s; sSq[wave * N + c] = q; }
        }
    }
    if constexpr (STATS) {
        __syncthreads();
        for (int c = tid; c < N; c += 128) {
            float s = sSum[c] + sSum[N + c] + sSum[2 * N + c] + sSum[3 * N + c];
            float q = sSq[c]  + sSq[N + c]  + sSq[2 * N + c]  + sSq[3 * N + c];
            partials[(size_t)blockIdx.x * (2 * N) + c]     = s;
            partials[(size_t)blockIdx.x * (2 * N) + N + c] = q;
        }
    }
}

// ------------------------- BN finalize: fold mu/var/gamma/beta --------------
__global__ __launch_bounds__(128)
void bn_finalize_kernel(const float* __restrict__ partials, int nblk,
                        const float* __restrict__ gamma,
                        const float* __restrict__ beta,
                        float* __restrict__ scaleShift, float invB)
{
    int c = threadIdx.x;                       // 128 features
    float s = 0.0f, q = 0.0f;
    for (int b = 0; b < nblk; ++b) {
        s += partials[(size_t)b * 256 + c];
        q += partials[(size_t)b * 256 + 128 + c];
    }
    float mu   = s * invB;
    float var  = q * invB - mu * mu;
    float rstd = rsqrtf(var + 1e-5f);
    float sc   = gamma[c] * rstd;
    scaleShift[c]       = sc;
    scaleShift[128 + c] = beta[c] - mu * sc;
}

// -------------------------------- coupling 1 --------------------------------
// x22 = x12*exp(s1)+t1 ; y odd cols ; build in2 f16 ; partial sum of s1.
__global__ __launch_bounds__(256)
void couple1_kernel(const float* __restrict__ x, const float* __restrict__ cond,
                    const float* __restrict__ S, const float* __restrict__ T,
                    _Float16* __restrict__ in16, float* __restrict__ y,
                    float* __restrict__ part)
{
    __shared__ float red[256];
    size_t idx = (size_t)blockIdx.x * 256 + threadIdx.x;  // B*96 total
    int row = (int)(idx / 96), j = (int)(idx % 96);
    float contrib = 0.0f;
    if (j < 48) {
        float s1 = atanf(S[(size_t)row*48 + j] * (1.0f / SFc)) * (2.0f * SFc / PI_F);
        float x12 = x[(size_t)row*DD + 2*j + 1] + sinf((float)(2*j + 1));
        float x22 = x12 * expf(s1) + T[(size_t)row*48 + j];
        y[(size_t)row*DD + 2*j + 1] = x22;
        in16[idx] = (_Float16)x22;
        contrib = s1;
    } else if (j < 80) {
        in16[idx] = (_Float16)cond[(size_t)row*CC + (j - 48)];
    } else {
        in16[idx] = (_Float16)0.0f;
    }
    red[threadIdx.x] = contrib;
    __syncthreads();
    for (int s = 128; s > 0; s >>= 1) {
        if (threadIdx.x < s) red[threadIdx.x] += red[threadIdx.x + s];
        __syncthreads();
    }
    if (threadIdx.x == 0) part[blockIdx.x] = red[0];
}

// -------------------------------- coupling 2 --------------------------------
__global__ __launch_bounds__(256)
void couple2_kernel(const float* __restrict__ x, const float* __restrict__ S,
                    const float* __restrict__ T, float* __restrict__ y,
                    float* __restrict__ part)
{
    __shared__ float red[256];
    size_t idx = (size_t)blockIdx.x * 256 + threadIdx.x;  // B*48 total
    int row = (int)(idx / 48), j = (int)(idx % 48);
    float s2 = atanf(S[idx] * (1.0f / SFc)) * (2.0f * SFc / PI_F);
    float x11 = x[(size_t)row*DD + 2*j] + sinf((float)(2*j));
    y[(size_t)row*DD + 2*j] = x11 * expf(s2) + T[idx];
    red[threadIdx.x] = s2;
    __syncthreads();
    for (int s = 128; s > 0; s >>= 1) {
        if (threadIdx.x < s) red[threadIdx.x] += red[threadIdx.x + s];
        __syncthreads();
    }
    if (threadIdx.x == 0) part[blockIdx.x] = red[0];
}

// --------------------------- log-det final reduce ---------------------------
__global__ __launch_bounds__(256)
void logdet_final_kernel(const float* __restrict__ p1, int n1,
                         const float* __restrict__ p2, int n2,
                         float* __restrict__ out, float invB)
{
    __shared__ float red[256];
    float s = 0.0f;
    for (int i = threadIdx.x; i < n1; i += 256) s += p1[i];
    for (int i = threadIdx.x; i < n2; i += 256) s += p2[i];
    red[threadIdx.x] = s;
    __syncthreads();
    for (int t = 128; t > 0; t >>= 1) {
        if (threadIdx.x < t) red[threadIdx.x] += red[threadIdx.x + t];
        __syncthreads();
    }
    if (threadIdx.x == 0) out[0] = red[0] * invB;
}

// ---------------------------------------------------------------------------
extern "C" void kernel_launch(void* const* d_in, const int* in_sizes, int n_in,
                              void* d_out, int out_size, void* d_ws, size_t ws_size,
                              hipStream_t stream)
{
    (void)in_sizes; (void)n_in; (void)out_size; (void)ws_size;
    const float* x    = (const float*)d_in[0];
    const float* cond = (const float*)d_in[1];
    const float* W0   = (const float*)d_in[2];
    const float* b0   = (const float*)d_in[3];
    const float* g0   = (const float*)d_in[4];
    const float* be0  = (const float*)d_in[5];
    const float* Ws   = (const float*)d_in[6];
    const float* bs   = (const float*)d_in[7];
    const float* gs   = (const float*)d_in[8];
    const float* bes  = (const float*)d_in[9];
    const float* Wf   = (const float*)d_in[10];
    const float* bf   = (const float*)d_in[11];
    float* y = (float*)d_out;

    char* ws = (char*)d_ws;
    _Float16* wt   = (_Float16*)(ws + OFF_WT);
    _Float16* in16 = (_Float16*)(ws + OFF_IN16);
    float* hA      = (float*)(ws + OFF_HA);
    float* hB      = (float*)(ws + OFF_HB);
    float* Sb      = (float*)(ws + OFF_S);
    float* Tb      = (float*)(ws + OFF_T);
    float* parts   = (float*)(ws + OFF_PART);
    float* ss      = (float*)(ws + OFF_SS);
    float* ld1     = (float*)(ws + OFF_LD1);
    float* ld2     = (float*)(ws + OFF_LD2);

    const float invB = 1.0f / (float)BATCH;

    // 1. weights -> f16 transposed
    prep_weights_kernel<<<((int)WT_ELEMS + 255) / 256, 256, 0, stream>>>(W0, Ws, Wf, wt);

    // 2. in1 = [x_even + pe_even, cond, pad] as f16
    build_in1_kernel<<<BATCH * 96 / 256, 256, 0, stream>>>(x, cond, in16);

    // one 5-layer net: in16 (f16,K=96) -> out (f32 [B][48])
    auto run_net = [&](int n, float* outBuf) {
        // L0: 96(padded 80) -> 128
        gemm_bn_kernel<96, 128, false, true><<<GEMM_GRID, 128, 0, stream>>>(
            in16, wt + (size_t)n * 12288, b0 + n * 128, nullptr, hA, parts);
        bn_finalize_kernel<<<1, 128, 0, stream>>>(parts, GEMM_GRID,
            g0 + n * 128, be0 + n * 128, ss, invB);
        // L1..L3: 128 -> 128 (BN+lrelu fused into A-load)
        float* hcur = hA; float* hnext = hB;
        for (int j = 0; j < 3; ++j) {
            gemm_bn_kernel<128, 128, true, true><<<GEMM_GRID, 128, 0, stream>>>(
                hcur, wt + 49152 + (size_t)(n * 3 + j) * 16384,
                bs + (n * 3 + j) * 128, ss, hnext, parts);
            bn_finalize_kernel<<<1, 128, 0, stream>>>(parts, GEMM_GRID,
                gs + (n * 3 + j) * 128, bes + (n * 3 + j) * 128, ss, invB);
            float* tmp = hcur; hcur = hnext; hnext = tmp;
        }
        // final: 128 -> 48 (BN+lrelu of L3 fused into A-load)
        gemm_bn_kernel<128, 48, true, false><<<GEMM_GRID, 128, 0, stream>>>(
            hcur, wt + 245760 + (size_t)n * 6144, bf + n * 48, ss, outBuf, nullptr);
    };

    // 3. pass 1: nets 0 (scale) and 1 (translate), then coupling
    run_net(0, Sb);
    run_net(1, Tb);
    couple1_kernel<<<BATCH * 96 / 256, 256, 0, stream>>>(x, cond, Sb, Tb, in16, y, ld1);

    // 4. pass 2: nets 2 and 3 on in2, then coupling
    run_net(2, Sb);
    run_net(3, Tb);
    couple2_kernel<<<BATCH * 48 / 256, 256, 0, stream>>>(x, Sb, Tb, y, ld2);

    // 5. log-det scalar
    logdet_final_kernel<<<1, 256, 0, stream>>>(ld1, BATCH * 96 / 256,
                                               ld2, BATCH * 48 / 256,
                                               y + (size_t)BATCH * DD, invB);
}
